// TemporalAttention_7945689497974
// MI455X (gfx1250) — compile-verified, matrix operand order checked by reading, not run on hardware
//
#include <hip/hip_runtime.h>

// ---------------------------------------------------------------------------
// Types for gfx1250 WMMA  (v_wmma_f32_16x16x32_f16)
// ---------------------------------------------------------------------------
typedef _Float16 v8h  __attribute__((ext_vector_type(8)));
typedef _Float16 v16h __attribute__((ext_vector_type(16)));
typedef float    v8f  __attribute__((ext_vector_type(8)));
typedef int      v4i_ __attribute__((vector_size(16)));   // matches builtin param

#define B_ 32
#define T_ 1024
#define D_ 512
#define H_ 256

#ifndef __has_builtin
#define __has_builtin(x) 0
#endif
#if __has_builtin(__builtin_amdgcn_global_load_async_to_lds_b128)
#define HAVE_ASYNC_LDS 1
#else
#define HAVE_ASYNC_LDS 0
#endif

// 16-byte global -> LDS copy. Async (ASYNCcnt-tracked, bypasses VGPRs) when
// the gfx1250 builtin exists; plain VGPR staging otherwise.
__device__ __forceinline__ void cp16_g2l(_Float16* ldst, const _Float16* gsrc) {
#if HAVE_ASYNC_LDS
    __builtin_amdgcn_global_load_async_to_lds_b128(
        (__attribute__((address_space(1))) v4i_*)gsrc,
        (__attribute__((address_space(3))) v4i_*)ldst, 0, 0);
#else
    *(v8h*)ldst = *(const v8h*)gsrc;
#endif
}

template <int N>
__device__ __forceinline__ void wait_async() {
#if __has_builtin(__builtin_amdgcn_s_wait_asynccnt)
    __builtin_amdgcn_s_wait_asynccnt(N);
#elif HAVE_ASYNC_LDS
    asm volatile("s_wait_asynccnt %0" :: "n"(N) : "memory");
#endif
}

// Build a 16-element f16 fragment: lanes 0-15 take K {0..7,16..23},
// lanes 16-31 take K {8..15,24..31}; p already offset by (lane<16?0:8).
__device__ __forceinline__ v16h mk_frag(const _Float16* p) {
    v8h lo = *(const v8h*)p;
    v8h hi = *(const v8h*)(p + 16);
    return __builtin_shufflevector(lo, hi, 0,1,2,3,4,5,6,7,8,9,10,11,12,13,14,15);
}

__device__ __forceinline__ v8f wmma16(v16h a, v16h b, v8f c) {
    return __builtin_amdgcn_wmma_f32_16x16x32_f16(
        false, a, false, b, (short)0, c, false, false);
}

// ---------------------------------------------------------------------------
// Kernel 0: zero the pooled-ctx accumulator
// ---------------------------------------------------------------------------
__global__ void ta_zero_kernel(float* __restrict__ p, int n) {
    int i = blockIdx.x * 256 + threadIdx.x;
    if (i < n) p[i] = 0.0f;
}

// ---------------------------------------------------------------------------
// Kernel 1: fused QKV projection. out[m,n] = sum_k x[m,k]*W[k,n]
// M=32768, K=512, N=256; blockIdx.y selects Wq/Wk/Wv. 64x256 tile / block.
// Register-prefetch of next K-tile overlaps global loads with WMMA.
// ---------------------------------------------------------------------------
__global__ __launch_bounds__(256)
void ta_qkv_kernel(const float* __restrict__ x,
                   const float* __restrict__ Wq,
                   const float* __restrict__ Wk,
                   const float* __restrict__ Wv,
                   _Float16*    __restrict__ qkv)   // [3][B*T][H] f16
{
    __shared__ __align__(16) _Float16 sA [64 * 32];   // x tile   [m][k]
    __shared__ __align__(16) _Float16 sBT[256 * 32];  // W^T tile [n][k]

    const int tid  = threadIdx.x;
    const int wave = tid >> 5;
    const int lane = tid & 31;
    const int m0   = blockIdx.x * 64;
    const int which = blockIdx.y;

    const float* W = (which == 0) ? Wq : ((which == 1) ? Wk : Wv);
    _Float16* out = qkv + (size_t)which * ((size_t)B_ * T_ * H_);

    const int mbase = (wave & 3) * 16;
    const int nhalf = (wave >> 2) * 128;
    const int koff  = (lane < 16) ? 0 : 8;
    const int lan16 = lane & 15;

    const v8f vzero = {0.f,0.f,0.f,0.f,0.f,0.f,0.f,0.f};
    v8f acc[8];
#pragma unroll
    for (int i = 0; i < 8; ++i) acc[i] = vzero;

    // staging registers
    const int row = tid >> 2;
    const int kc  = (tid & 3) * 8;
    float4 xf0, xf1;
    v8h w0, w1, w2, w3;

    auto load_x = [&](int k0) {
        const float* src = x + (size_t)(m0 + row) * D_ + k0 + kc;
        xf0 = *(const float4*)src;
        xf1 = *(const float4*)(src + 4);
    };
    auto load_w = [&](int k0) {
#pragma unroll
        for (int i = 0; i < 8; ++i) w0[i] = (_Float16)W[(size_t)(k0 + i)      * H_ + tid];
#pragma unroll
        for (int i = 0; i < 8; ++i) w1[i] = (_Float16)W[(size_t)(k0 + 8 + i)  * H_ + tid];
#pragma unroll
        for (int i = 0; i < 8; ++i) w2[i] = (_Float16)W[(size_t)(k0 + 16 + i) * H_ + tid];
#pragma unroll
        for (int i = 0; i < 8; ++i) w3[i] = (_Float16)W[(size_t)(k0 + 24 + i) * H_ + tid];
    };

    load_x(0);
    load_w(0);

    for (int kt = 0; kt < 16; ++kt) {
        __syncthreads();
        // ---- commit staged tile to LDS ----
        {
            v8h h8;
            h8[0]=(_Float16)xf0.x; h8[1]=(_Float16)xf0.y; h8[2]=(_Float16)xf0.z; h8[3]=(_Float16)xf0.w;
            h8[4]=(_Float16)xf1.x; h8[5]=(_Float16)xf1.y; h8[6]=(_Float16)xf1.z; h8[7]=(_Float16)xf1.w;
            *(v8h*)(sA + row * 32 + kc) = h8;
            *(v8h*)(sBT + tid * 32 +  0) = w0;
            *(v8h*)(sBT + tid * 32 +  8) = w1;
            *(v8h*)(sBT + tid * 32 + 16) = w2;
            *(v8h*)(sBT + tid * 32 + 24) = w3;
        }
        // ---- prefetch next tile (overlaps the WMMA section below) ----
        if (kt + 1 < 16) {
            load_x((kt + 1) * 32);
            load_w((kt + 1) * 32);
        }
        __syncthreads();

        // ---- WMMA: 1 A-frag reused across 8 N-chunks ----
        v16h af = mk_frag(sA + (mbase + lan16) * 32 + koff);
#pragma unroll
        for (int nc = 0; nc < 8; ++nc) {
            int nbase = nhalf + nc * 16;
            v16h bf = mk_frag(sBT + (nbase + lan16) * 32 + koff);
            acc[nc] = wmma16(af, bf, acc[nc]);
        }
    }

    // ---- store f16 output ----
#pragma unroll
    for (int nc = 0; nc < 8; ++nc) {
        int n = nhalf + nc * 16 + lan16;
#pragma unroll
        for (int r = 0; r < 8; ++r) {
            int m = m0 + mbase + r + 8 * (lane >> 4);
            out[(size_t)m * H_ + n] = (_Float16)acc[nc][r];
        }
    }
}

// ---------------------------------------------------------------------------
// Kernel 2: flash attention + masked pooling of ctx into pooled_ctx[b][h].
// Block = 256 thr (8 waves); each wave owns a 16-row query tile; all waves
// share the K/V^T LDS tiles.  K tile is double-buffered and filled with
// async global->LDS copies; V tile is register-prefetched (needs transpose).
// grid = (T/128, B).
// ---------------------------------------------------------------------------
__global__ __launch_bounds__(256)
void ta_attn_kernel(const _Float16* __restrict__ q,
                    const _Float16* __restrict__ k,
                    const _Float16* __restrict__ v,
                    const unsigned char* __restrict__ mask,
                    float* __restrict__ pooled_ctx)   // [B][H]
{
    __shared__ __align__(16) _Float16 sK [2][32 * H_];   // [buf][keyRel][h] 32KB
    __shared__ __align__(16) _Float16 sVT[H_ * 32];      // [h][keyRel]      16KB
    __shared__ __align__(16) _Float16 sP [8][16 * 32];   // per-wave P        8KB

    const int tid  = threadIdx.x;
    const int wave = tid >> 5;
    const int lane = tid & 31;
    const int b    = blockIdx.y;
    const int q0   = blockIdx.x * 128 + wave * 16;

    const _Float16* qb = q + (size_t)b * T_ * H_;
    const _Float16* kb = k + (size_t)b * T_ * H_;
    const _Float16* vb = v + (size_t)b * T_ * H_;
    const unsigned char* mb = mask + (size_t)b * T_;

    const int koff  = (lane < 16) ? 0 : 8;
    const int lan16 = lane & 15;
    const int kr    = tid >> 3;          // staging: key row 0..31
    const int hc    = (tid & 7) * 32;    // staging: h chunk

    // ---- preload q A-fragments (8 chunks of K=32 over H=256) ----
    v16h qf[8];
    {
        const _Float16* p = qb + (size_t)(q0 + lan16) * H_ + koff;
#pragma unroll
        for (int c = 0; c < 8; ++c) qf[c] = mk_frag(p + c * 32);
    }

    const v8f vzero = {0.f,0.f,0.f,0.f,0.f,0.f,0.f,0.f};
    v8f O[16];
#pragma unroll
    for (int i = 0; i < 16; ++i) O[i] = vzero;
    float mrow[8], lrow[8];
#pragma unroll
    for (int r = 0; r < 8; ++r) { mrow[r] = -1e30f; lrow[r] = 0.f; }

    // ---- prologue: async K tile 0 -> buf0; V tile 0 -> regs ----
    {
        const _Float16* ks = kb + (size_t)kr * H_ + hc;
        _Float16* kd = &sK[0][kr * H_ + hc];
#pragma unroll
        for (int i = 0; i < 4; ++i) cp16_g2l(kd + i * 8, ks + i * 8);
    }
    v8h vst[4];
    {
        const _Float16* vs = vb + (size_t)kr * H_ + hc;
#pragma unroll
        for (int i = 0; i < 4; ++i) vst[i] = *(const v8h*)(vs + i * 8);
    }

    for (int t = 0; t < T_ / 32; ++t) {
        const int s0 = t * 32;
        __syncthreads();   // (A) everyone done reading sVT and sK[(t+1)&1]

        // issue async K copies for tile t+1 into the other buffer
        if (t + 1 < T_ / 32) {
            const _Float16* ks = kb + (size_t)(s0 + 32 + kr) * H_ + hc;
            _Float16* kd = &sK[(t + 1) & 1][kr * H_ + hc];
#pragma unroll
            for (int i = 0; i < 4; ++i) cp16_g2l(kd + i * 8, ks + i * 8);
        }
        // commit V tile t, transposed
#pragma unroll
        for (int i = 0; i < 4; ++i)
#pragma unroll
            for (int j = 0; j < 8; ++j)
                sVT[(hc + i * 8 + j) * 32 + kr] = vst[i][j];
        // prefetch V tile t+1 (overlaps WMMA below)
        if (t + 1 < T_ / 32) {
            const _Float16* vs = vb + (size_t)(s0 + 32 + kr) * H_ + hc;
#pragma unroll
            for (int i = 0; i < 4; ++i) vst[i] = *(const v8h*)(vs + i * 8);
        }
        // tile t's 4 async copies (in-order ASYNCcnt) must have landed
        if (t + 1 < T_ / 32) wait_async<4>(); else wait_async<0>();
        __syncthreads();   // (B)

        const _Float16* kt_ = &sK[t & 1][0];

        // ---- scores S = q . k^T for 32 keys ----
        v8f S0 = vzero, S1 = vzero;
#pragma unroll
        for (int c = 0; c < 8; ++c) {
            v16h bf0 = mk_frag(kt_ + lan16        * H_ + c * 32 + koff);
            v16h bf1 = mk_frag(kt_ + (lan16 + 16) * H_ + c * 32 + koff);
            S0 = wmma16(qf[c], bf0, S0);
            S1 = wmma16(qf[c], bf1, S1);
        }

        const float scale = 0.0625f;   // 1/sqrt(256)
        float mk0 = mb[s0 + lan16]      ? 0.f : -1e30f;
        float mk1 = mb[s0 + 16 + lan16] ? 0.f : -1e30f;

        float p0v[8], p1v[8], alpha[8];
#pragma unroll
        for (int r = 0; r < 8; ++r) {
            p0v[r] = S0[r] * scale + mk0;
            p1v[r] = S1[r] * scale + mk1;
        }
        // ---- online softmax; row reductions across 16-lane halves ----
#pragma unroll
        for (int r = 0; r < 8; ++r) {
            float mx = fmaxf(p0v[r], p1v[r]);
#pragma unroll
            for (int i = 1; i < 16; i <<= 1) mx = fmaxf(mx, __shfl_xor(mx, i, 32));
            float mnew = fmaxf(mrow[r], mx);
            alpha[r] = __expf(mrow[r] - mnew);
            mrow[r]  = mnew;
            p0v[r] = __expf(p0v[r] - mnew);
            p1v[r] = __expf(p1v[r] - mnew);
            float rs = p0v[r] + p1v[r];
#pragma unroll
            for (int i = 1; i < 16; i <<= 1) rs += __shfl_xor(rs, i, 32);
            lrow[r] = lrow[r] * alpha[r] + rs;
        }
#pragma unroll
        for (int h = 0; h < 16; ++h)
#pragma unroll
            for (int r = 0; r < 8; ++r) O[h][r] *= alpha[r];

        // ---- restage P (C layout -> A layout) through LDS ----
        _Float16* pw = &sP[wave][0];
#pragma unroll
        for (int r = 0; r < 8; ++r) {
            int m = r + 8 * (lane >> 4);
            pw[m * 32 + lan16]      = (_Float16)p0v[r];
            pw[m * 32 + 16 + lan16] = (_Float16)p1v[r];
        }
        __syncthreads();   // (C)

        // ---- ctx += P(16x32) x V(32xH): 16 WMMAs over H chunks ----
        v16h pf = mk_frag(pw + lan16 * 32 + koff);
#pragma unroll
        for (int hcn = 0; hcn < 16; ++hcn) {
            v16h bfv = mk_frag(sVT + (hcn * 16 + lan16) * 32 + koff);
            O[hcn] = wmma16(pf, bfv, O[hcn]);
        }
    }

    // ---- finalize: apply 1/l and query mask, pool the 16 rows ----
    float winv[8];
#pragma unroll
    for (int r = 0; r < 8; ++r) {
        int m = r + 8 * (lane >> 4);
        float qm = mb[q0 + m] ? 1.f : 0.f;
        winv[r] = qm / fmaxf(lrow[r], 1e-20f);
    }
#pragma unroll
    for (int hcn = 0; hcn < 16; ++hcn) {
        float s = 0.f;
#pragma unroll
        for (int r = 0; r < 8; ++r) s += O[hcn][r] * winv[r];
        s += __shfl_xor(s, 16, 32);
        if (lane < 16)
            atomicAdd(pooled_ctx + b * H_ + hcn * 16 + lan16, s);
    }
}

// ---------------------------------------------------------------------------
// Kernel 3: pooled = (pooled_ctx @ Wo) / denom.  [32,256]x[256,512] — tiny.
// ---------------------------------------------------------------------------
__global__ __launch_bounds__(256)
void ta_outproj_kernel(const float* __restrict__ pc,          // [B][H]
                       const float* __restrict__ Wo,          // [H][D]
                       const unsigned char* __restrict__ mask,
                       float* __restrict__ out)               // [B][D]
{
    __shared__ float s_pc [H_];
    __shared__ float s_red[256];
    const int b = blockIdx.y;
    const int t = threadIdx.x;
    const int d = blockIdx.x * 256 + t;

    float cnt = 0.f;
#pragma unroll
    for (int i = 0; i < 4; ++i) cnt += mask[b * T_ + t + i * 256] ? 1.f : 0.f;
    s_red[t] = cnt;
    s_pc[t]  = pc[b * H_ + t];
    __syncthreads();
    for (int off = 128; off > 0; off >>= 1) {
        if (t < off) s_red[t] += s_red[t + off];
        __syncthreads();
    }
    const float denom = fmaxf(s_red[0], 1.f);

    float sum = 0.f;
#pragma unroll 8
    for (int h = 0; h < H_; ++h) sum += s_pc[h] * Wo[h * D_ + d];
    out[b * D_ + d] = sum / denom;
}

// ---------------------------------------------------------------------------
// Launch
// ---------------------------------------------------------------------------
extern "C" void kernel_launch(void* const* d_in, const int* in_sizes, int n_in,
                              void* d_out, int out_size, void* d_ws, size_t ws_size,
                              hipStream_t stream) {
    const float* x            = (const float*)d_in[0];
    const unsigned char* mask = (const unsigned char*)d_in[1];
    const float* Wq           = (const float*)d_in[2];
    const float* Wk           = (const float*)d_in[3];
    const float* Wv           = (const float*)d_in[4];
    const float* Wo           = (const float*)d_in[5];
    float* out                = (float*)d_out;

    const size_t elems = (size_t)B_ * T_ * H_;             // 8,388,608 per tensor
    _Float16* qkv = (_Float16*)d_ws;                       // 3 * 16 MB f16
    float* pc = (float*)((char*)d_ws + 3 * elems * sizeof(_Float16));

    ta_zero_kernel<<<dim3((B_ * H_ + 255) / 256), dim3(256), 0, stream>>>(pc, B_ * H_);
    ta_qkv_kernel<<<dim3((B_ * T_) / 64, 3), dim3(256), 0, stream>>>(x, Wq, Wk, Wv, qkv);
    ta_attn_kernel<<<dim3(T_ / 128, B_), dim3(256), 0, stream>>>(
        qkv, qkv + elems, qkv + 2 * elems, mask, pc);
    ta_outproj_kernel<<<dim3(D_ / 256, B_), dim3(256), 0, stream>>>(pc, Wo, mask, out);
}
